// SelfAttention_80685255623127
// MI455X (gfx1250) — matrix operand order repaired and checked by reading, not compiled, not run on hardware
//
#include <hip/hip_runtime.h>
#include <hip/hip_bf16.h>

typedef _Float16 v16h __attribute__((ext_vector_type(16)));
typedef _Float16 v8h  __attribute__((ext_vector_type(8)));
typedef float    v8f  __attribute__((ext_vector_type(8)));
typedef float    v4f  __attribute__((ext_vector_type(4)));
typedef int      v4i  __attribute__((ext_vector_type(4)));

#define EMBED 512
#define NH    8
#define HD    64
#define BB    2
#define SS    4096
#define NTOK  (BB*SS)              // 8192 tokens
#define ELEMS ((size_t)NTOK*EMBED) // 4,194,304 halves per tensor

// log2(e)/sqrt(HD): folded into Q at projection time so attention scores
// come out of the WMMA already in the exp2 domain.
#define SC_Q  0.1803368801111137f
// -10000 * log2(e): additive mask bias in the exp2 domain.
#define MSK_B (-14426.950408889634f)

__device__ __forceinline__ v16h cat8(v8h a, v8h b) {
    v16h r;
#pragma unroll
    for (int i = 0; i < 8; ++i) { r[i] = a[i]; r[8 + i] = b[i]; }
    return r;
}

// ---------------------------------------------------------------------------
// Kernel 1: fused QKV projection. y[t,o] = sum_e x[t,e]*W[o,e] + b[o]
// One wave computes one 16(t) x 16(o) tile with 16 WMMAs over e=512.
// Q (pre-scaled by SC_Q), K stored f16 row-major [b,h,s,64];
// V stored f16 transposed [b,h,64,s] so attention B-fragments are contiguous.
// grid = (2048, 3), block = 256 (8 waves) -> exactly 512*32 tiles per matrix.
// ---------------------------------------------------------------------------
__global__ __launch_bounds__(256) void qkv_proj_kernel(
    const float* __restrict__ x,
    const float* __restrict__ Wq, const float* __restrict__ bq,
    const float* __restrict__ Wk, const float* __restrict__ bk,
    const float* __restrict__ Wv, const float* __restrict__ bv,
    _Float16* __restrict__ Qh, _Float16* __restrict__ Kh,
    _Float16* __restrict__ Vt)
{
    const int lane = threadIdx.x & 31;
    const int wid  = threadIdx.x >> 5;
    const int mat  = blockIdx.y;                // 0=Q 1=K 2=V
    const int job  = blockIdx.x * 8 + wid;      // 0..16383
    const int tt   = job >> 5;                  // 0..511  (token tile)
    const int ot   = job & 31;                  // 0..31   (output tile)
    const int tbase = tt * 16;
    const int obase = ot * 16;
    const bool hi  = lane >= 16;
    const int l15  = lane & 15;

    const float* W    = (mat == 0) ? Wq : (mat == 1) ? Wk : Wv;
    const float* bias = (mat == 0) ? bq : (mat == 1) ? bk : bv;

    // A-frag source: row t = tbase+l15, K(e) offsets {0..7,16..23} lo / {8..15,24..31} hi
    const float* xrow = x + (size_t)(tbase + l15) * EMBED + (hi ? 8 : 0);
    // B-frag source: row o = obase+l15, K(e) offsets {0..7,16..23} lo / {8..15,24..31} hi
    const float* wrow = W + (size_t)(obase + l15) * EMBED + (hi ? 8 : 0);

    v8f acc = {};
    for (int e = 0; e < EMBED; e += 32) {
        v4f a0 = *(const v4f*)(xrow + e);
        v4f a1 = *(const v4f*)(xrow + e + 4);
        v4f a2 = *(const v4f*)(xrow + e + 16);
        v4f a3 = *(const v4f*)(xrow + e + 20);
        v16h af;
#pragma unroll
        for (int i = 0; i < 4; ++i) {
            af[i]      = (_Float16)a0[i];
            af[4 + i]  = (_Float16)a1[i];
            af[8 + i]  = (_Float16)a2[i];
            af[12 + i] = (_Float16)a3[i];
        }
        v4f b0 = *(const v4f*)(wrow + e);
        v4f b1 = *(const v4f*)(wrow + e + 4);
        v4f b2 = *(const v4f*)(wrow + e + 16);
        v4f b3 = *(const v4f*)(wrow + e + 20);
        v16h bf;
#pragma unroll
        for (int i = 0; i < 4; ++i) {
            bf[i]      = (_Float16)b0[i];
            bf[4 + i]  = (_Float16)b1[i];
            bf[8 + i]  = (_Float16)b2[i];
            bf[12 + i] = (_Float16)b3[i];
        }
        acc = __builtin_amdgcn_wmma_f32_16x16x32_f16(
            false, af, false, bf, (short)0, acc, false, false);
    }

    const float bv_   = bias[obase + l15];            // o fixed per lane
    const float scale = (mat == 0) ? SC_Q : 1.0f;     // fold score scaling into Q
#pragma unroll
    for (int v = 0; v < 8; ++v) acc[v] = (acc[v] + bv_) * scale;

    const int b  = tbase >> 12;                 // tbase / 4096
    const int s0 = tbase & (SS - 1);
    const int h  = obase >> 6;
    const int d0 = obase & 63;

    if (mat < 2) {
        _Float16* dst = (mat == 0 ? Qh : Kh) + (size_t)(b * NH + h) * SS * HD;
#pragma unroll
        for (int v = 0; v < 8; ++v) {
            const int s = s0 + v + (hi ? 8 : 0);
            dst[(size_t)s * HD + d0 + l15] = (_Float16)acc[v];
        }
    } else {
        _Float16* dst = Vt + (size_t)(b * NH + h) * HD * SS;
        const int d = d0 + l15;
#pragma unroll
        for (int v = 0; v < 8; ++v) {
            const int s = s0 + v + (hi ? 8 : 0);
            dst[(size_t)d * SS + s] = (_Float16)acc[v];   // contiguous over v -> b128
        }
    }
}

// ---------------------------------------------------------------------------
// Kernel 2: flash attention. One wave owns one (b,h, 16-query tile).
// Per 32-wide k-chunk: St = K_tile x Q^T (2 D-frags, already log2-scaled),
// mask via one fma, online softmax in registers (lane = q), P feeds the A
// operand with zero cross-lane movement, O(16x64) as 4 D-frags via P x Vt.
// Accumulator rescale is skipped wave-uniformly when no row max increased.
// grid = 512, block = 256 (8 waves) -> exactly 16*256 q-tiles.
// ---------------------------------------------------------------------------
__global__ __launch_bounds__(256) void attn_kernel(
    const _Float16* __restrict__ Qh, const _Float16* __restrict__ Kh,
    const _Float16* __restrict__ Vt, const int* __restrict__ mask,
    float* __restrict__ out)
{
    const int lane = threadIdx.x & 31;
    const int wid  = threadIdx.x >> 5;
    const int job  = blockIdx.x * 8 + wid;      // 0..4095
    const int bh   = job >> 8;                  // 0..15
    const int qt   = job & 255;
    const int qbase = qt * 16;
    const int b = bh >> 3, h = bh & 7;
    const bool hi  = lane >= 16;
    const int l15  = lane & 15;

    // Q as B-fragments (N = q across lanes, K = d): d {0..7,16..23} lo / {8..15,24..31} hi
    const _Float16* qrow = Qh + ((size_t)bh * SS + qbase + l15) * HD + (hi ? 8 : 0);
    const v16h qb0 = cat8(*(const v8h*)(qrow),      *(const v8h*)(qrow + 16));   // d 0..31
    const v16h qb1 = cat8(*(const v8h*)(qrow + 32), *(const v8h*)(qrow + 48));   // d 32..63

    const _Float16* kmat = Kh + (size_t)bh * SS * HD;
    const _Float16* vmat = Vt + (size_t)bh * HD * SS;
    const int* mrow = mask + (size_t)(qbase + l15) * SS; // lane's query row
    const int koff  = hi ? 8 : 0;

    v8f o0 = {}, o1 = {}, o2 = {}, o3 = {};
    float m_run = -3.0e38f, l_run = 0.0f;

    for (int kb = 0; kb < SS; kb += 32) {
        // prefetch next chunk (wrapped so addresses stay in-buffer)
        {
            const int kn = (kb + 32) & (SS - 1);
            __builtin_prefetch(kmat + (size_t)(kn + l15) * HD, 0, 3);
            __builtin_prefetch(vmat + (size_t)l15 * SS + kn, 0, 3);
            __builtin_prefetch(mrow + kn, 0, 3);
        }

        // K A-fragments: rows kb..kb+15 (st0) and kb+16..kb+31 (st1)
        const _Float16* kr0 = kmat + (size_t)(kb + l15) * HD + koff;
        const _Float16* kr1 = kmat + (size_t)(kb + 16 + l15) * HD + koff;
        const v16h kf00 = cat8(*(const v8h*)(kr0),      *(const v8h*)(kr0 + 16)); // d0..31
        const v16h kf01 = cat8(*(const v8h*)(kr0 + 32), *(const v8h*)(kr0 + 48)); // d32..63
        const v16h kf10 = cat8(*(const v8h*)(kr1),      *(const v8h*)(kr1 + 16));
        const v16h kf11 = cat8(*(const v8h*)(kr1 + 32), *(const v8h*)(kr1 + 48));

        v8f z = {};
        v8f st0 = __builtin_amdgcn_wmma_f32_16x16x32_f16(false, kf00, false, qb0, (short)0, z,   false, false);
        st0     = __builtin_amdgcn_wmma_f32_16x16x32_f16(false, kf01, false, qb1, (short)0, st0, false, false);
        v8f st1 = __builtin_amdgcn_wmma_f32_16x16x32_f16(false, kf10, false, qb0, (short)0, z,   false, false);
        st1     = __builtin_amdgcn_wmma_f32_16x16x32_f16(false, kf11, false, qb1, (short)0, st1, false, false);

        // mask: lane's row q, k = kb + {0..7|8..15} (st0) and +16 (st1)
        const v4i ma = *(const v4i*)(mrow + kb + koff);
        const v4i mb = *(const v4i*)(mrow + kb + koff + 4);
        const v4i mc = *(const v4i*)(mrow + kb + koff + 16);
        const v4i md = *(const v4i*)(mrow + kb + koff + 20);

        float s0v[8], s1v[8];
#pragma unroll
        for (int v = 0; v < 4; ++v) {
            s0v[v]     = fmaf((float)ma[v], MSK_B, st0[v]);
            s0v[4 + v] = fmaf((float)mb[v], MSK_B, st0[4 + v]);
            s1v[v]     = fmaf((float)mc[v], MSK_B, st1[v]);
            s1v[4 + v] = fmaf((float)md[v], MSK_B, st1[4 + v]);
        }

        // online softmax (lane holds 16 of its row's 32 values; pair with lane^16)
        float cmax = s0v[0];
#pragma unroll
        for (int v = 1; v < 8; ++v) cmax = fmaxf(cmax, s0v[v]);
#pragma unroll
        for (int v = 0; v < 8; ++v) cmax = fmaxf(cmax, s1v[v]);
        cmax = fmaxf(cmax, __shfl_xor(cmax, 16, 32));
        const float m_new = fmaxf(m_run, cmax);

        float p0[8], p1[8], psum = 0.0f;
#pragma unroll
        for (int v = 0; v < 8; ++v) {
            p0[v] = exp2f(s0v[v] - m_new);
            p1[v] = exp2f(s1v[v] - m_new);
            psum += p0[v] + p1[v];
        }
        psum += __shfl_xor(psum, 16, 32);

        if (__any(cmax > m_run)) {
            // some row got a new max: rescale accumulators (wave-uniform branch,
            // EXEC untouched -> WMMA-safe)
            const float corr = exp2f(m_run - m_new);
            l_run = l_run * corr + psum;
            m_run = m_new;
#pragma unroll
            for (int v = 0; v < 8; ++v) {
                const float f = __shfl(corr, v + (hi ? 8 : 0), 32);
                o0[v] *= f; o1[v] *= f; o2[v] *= f; o3[v] *= f;
            }
        } else {
            l_run += psum;   // corr == 1 for every lane
        }

        // P fragment: D-layout of St == A-layout of P, no cross-lane movement
        v16h pf;
#pragma unroll
        for (int v = 0; v < 8; ++v) {
            pf[v]     = (_Float16)p0[v];
            pf[8 + v] = (_Float16)p1[v];
        }

        // V B-fragments from transposed V: lane's column d, k {0..7,16..23} lo / {8..15,24..31} hi
        const _Float16* vr = vmat + (size_t)l15 * SS + kb + (hi ? 8 : 0);
        o0 = __builtin_amdgcn_wmma_f32_16x16x32_f16(false, pf, false, cat8(*(const v8h*)(vr),           *(const v8h*)(vr + 16)),           (short)0, o0, false, false);
        o1 = __builtin_amdgcn_wmma_f32_16x16x32_f16(false, pf, false, cat8(*(const v8h*)(vr + 16 * SS), *(const v8h*)(vr + 16 * SS + 16)), (short)0, o1, false, false);
        o2 = __builtin_amdgcn_wmma_f32_16x16x32_f16(false, pf, false, cat8(*(const v8h*)(vr + 32 * SS), *(const v8h*)(vr + 32 * SS + 16)), (short)0, o2, false, false);
        o3 = __builtin_amdgcn_wmma_f32_16x16x32_f16(false, pf, false, cat8(*(const v8h*)(vr + 48 * SS), *(const v8h*)(vr + 48 * SS + 16)), (short)0, o3, false, false);
    }

    const float inv = 1.0f / l_run;   // l_run >= 1 (max element contributes 1)
#pragma unroll
    for (int v = 0; v < 8; ++v) {
        const float f = __shfl(inv, v + (hi ? 8 : 0), 32);
        const int q = qbase + v + (hi ? 8 : 0);
        const size_t base = (size_t)(b * SS + q) * EMBED + (size_t)h * HD + l15;
        out[base]      = o0[v] * f;
        out[base + 16] = o1[v] * f;
        out[base + 32] = o2[v] * f;
        out[base + 48] = o3[v] * f;
    }
}

extern "C" void kernel_launch(void* const* d_in, const int* in_sizes, int n_in,
                              void* d_out, int out_size, void* d_ws, size_t ws_size,
                              hipStream_t stream) {
    const float* x    = (const float*)d_in[0];
    const int*   mask = (const int*)d_in[1];
    const float* Wq   = (const float*)d_in[2];
    const float* bq   = (const float*)d_in[3];
    const float* Wk   = (const float*)d_in[4];
    const float* bk   = (const float*)d_in[5];
    const float* Wv   = (const float*)d_in[6];
    const float* bv   = (const float*)d_in[7];
    float* out = (float*)d_out;

    _Float16* Qh = (_Float16*)d_ws;
    _Float16* Kh = Qh + ELEMS;
    _Float16* Vt = Kh + ELEMS;   // 24 MB total f16 scratch

    dim3 g1(2048, 3);
    qkv_proj_kernel<<<g1, 256, 0, stream>>>(x, Wq, bq, Wk, bk, Wv, bv, Qh, Kh, Vt);
    attn_kernel<<<512, 256, 0, stream>>>(Qh, Kh, Vt, mask, out);
}
